// SSIMLoss_44495861186857
// MI455X (gfx1250) — matrix operand order repaired and checked by reading, not compiled, not run on hardware
//
#include <hip/hip_runtime.h>

// CDNA5 / gfx1250 SSIM loss: separable 11-tap Gaussian blurs as banded
// 16x16 = (16x32)x(32x16) matmuls on v_wmma_f32_16x16x32_f16, fused with the
// per-pixel SSIM map and a deterministic two-stage reduction.
//
// Round-2 changes: A operands (x,y) loaded from LDS once per tile and the five
// map operands derived with packed f16 muls; mid matrix stored column-major so
// spill/gather are b128 vector DS ops; next-tile halo prefetch.

typedef __attribute__((ext_vector_type(16))) _Float16 v16h;
typedef __attribute__((ext_vector_type(8)))  _Float16 v8h;
typedef __attribute__((ext_vector_type(8)))  float    v8f;

#define IMG_H   512
#define IMG_W   512
#define PLANES  96                                   // B*C = 32*3
#define TILES   (PLANES * (IMG_H/16) * (IMG_W/16))   // 98304 16x16 tiles
#define NWG     2048
#define WPG     8                                    // waves per 256-thread block (wave32)
#define SSIM_C1 6.5025f                              // (0.01*255)^2
#define SSIM_C2 58.5225f                             // (0.03*255)^2

#define SHUF16(lo, hi) \
  __builtin_shufflevector(lo, hi, 0, 1, 2, 3, 4, 5, 6, 7, 8, 9, 10, 11, 12, 13, 14, 15)

struct WaveLds {
  alignas(16) _Float16 xext[32][32];   // zero-padded 26x26 halo of (gen+1)/2
  alignas(16) _Float16 yext[32][32];   // zero-padded 26x26 halo of (ref+1)/2
  alignas(16) _Float16 midT[16][32];   // horizontal-pass result, column-major [n][k]
};                                      // 5 KB/wave -> 40 KB/block static LDS

__global__ __launch_bounds__(256, 1)
void ssim_tiles(const float* __restrict__ gen,
                const float* __restrict__ ref,
                const float* __restrict__ win,
                float* __restrict__ partial)
{
  __shared__ WaveLds lds[WPG];
  __shared__ float   wsum[WPG];

  const int lane = threadIdx.x & 31;
  const int wave = threadIdx.x >> 5;
  WaveLds&  L    = lds[wave];

  // 1-D taps from the separable 2-D window: w2d = g (x) g  =>  g[t] = sqrt(w2d[t][t]).
  float g[11];
  #pragma unroll
  for (int t = 0; t < 11; ++t) g[t] = sqrtf(win[t * 11 + t]);

  // Constant banded weight operands (built once, reused for every tile/map).
  // Wh: B operand (K=32 x N=16), Wh[k][n] = g[k-n] for 0<=k-n<=10.
  // 16-bit B layout: lane<16 -> col N=lane, elems j => K = j<8 ? j : 16+(j-8); lane>=16 -> N=lane-16, K = j<8 ? 8+j : 24+(j-8).
  v16h wh;
  {
    const int n  = lane & 15;
    const int kb = (lane >> 4) * 8;
    #pragma unroll
    for (int j = 0; j < 16; ++j) {
      const int d = ((j < 8) ? (kb + j) : (16 + kb + (j - 8))) - n;
      wh[j] = (_Float16)((d >= 0 && d <= 10) ? g[d] : 0.0f);
    }
  }
  // Wv: A operand (M=16 x K=32), Wv[m][k] = g[k-m] for 0<=k-m<=10.
  // 16-bit A layout: lane<16 -> row M=lane, elems j => K = j<8 ? j : 16+(j-8);
  //                  lane>=16 -> M=lane-16, K = j<8 ? 8+j : 24+(j-8).
  v16h wv;
  {
    const int m  = lane & 15;
    const int hf = lane >> 4;
    #pragma unroll
    for (int j = 0; j < 16; ++j) {
      const int k = (j < 8) ? (hf * 8 + j) : (16 + hf * 8 + (j - 8));
      const int d = k - m;
      wv[j] = (_Float16)((d >= 0 && d <= 10) ? g[d] : 0.0f);
    }
  }

  float acc = 0.0f;
  const int tstride = (int)gridDim.x * WPG;

  for (int tile = blockIdx.x * WPG + wave; tile < TILES; tile += tstride) {
    const int plane = tile >> 10;            // 1024 tiles per 512x512 plane
    const int t     = tile & 1023;
    const int r0    = (t >> 5) << 4;
    const int c0    = (t & 31) << 4;
    const float* p1 = gen + (size_t)plane * (IMG_H * IMG_W);
    const float* p2 = ref + (size_t)plane * (IMG_H * IMG_W);

    // Stage the 26x26 halo (zero padded to 32x32) as f16; coalesced 26-lane rows.
    const int  gc  = c0 - 5 + lane;
    const bool cok = (lane < 26) && (gc >= 0) && (gc < IMG_W);
    #pragma unroll 4
    for (int i = 0; i < 32; ++i) {
      const int gr = r0 - 5 + i;
      float xv = 0.0f, yv = 0.0f;
      if (cok && (i < 26) && (gr >= 0) && (gr < IMG_H)) {
        const size_t o = (size_t)gr * IMG_W + gc;
        xv = (p1[o] + 1.0f) * 0.5f;
        yv = (p2[o] + 1.0f) * 0.5f;
      }
      L.xext[i][lane] = (_Float16)xv;
      L.yext[i][lane] = (_Float16)yv;
    }

    // Prefetch next tile's halo rows (one line covers the 104B row span).
    {
      const int nt = tile + tstride;
      if (nt < TILES && lane < 26) {
        const int np  = nt >> 10;
        const int t2  = nt & 1023;
        const int nr  = ((t2 >> 5) << 4) - 5 + lane;
        const int ncc = ((t2 & 31) << 4) - 5;
        const int gr2 = nr < 0 ? 0 : (nr >= IMG_H ? IMG_H - 1 : nr);
        const int gc2 = ncc < 0 ? 0 : ncc;
        const size_t o = (size_t)np * (IMG_H * IMG_W) + (size_t)gr2 * IMG_W + gc2;
        __builtin_prefetch(&gen[o], 0, 1);
        __builtin_prefetch(&ref[o], 0, 1);
      }
    }

    // Load the two A-layout halves of x and y once; derive 5 maps in registers.
    v16h xa[2], ya[2];
    #pragma unroll
    for (int half = 0; half < 2; ++half) {
      const int m  = half * 16 + (lane & 15);
      const int kb = (lane >> 4) * 8;
      const v8h xlo = *(const v8h*)&L.xext[m][kb];
      const v8h xhi = *(const v8h*)&L.xext[m][kb + 16];
      const v8h ylo = *(const v8h*)&L.yext[m][kb];
      const v8h yhi = *(const v8h*)&L.yext[m][kb + 16];
      xa[half] = SHUF16(xlo, xhi);
      ya[half] = SHUF16(ylo, yhi);
    }

    v8f out[5];
    #pragma unroll
    for (int mp = 0; mp < 5; ++mp) {         // maps: x, y, x^2, y^2, xy
      // ---- horizontal pass: mid(32x16) = ext(32x32) * Wh(32x16), two M-halves
      v8f mid_d[2];
      #pragma unroll
      for (int half = 0; half < 2; ++half) {
        const v16h a = (mp == 0) ? xa[half]
                     : (mp == 1) ? ya[half]
                     : (mp == 2) ? xa[half] * xa[half]     // v_pk_mul_f16
                     : (mp == 3) ? ya[half] * ya[half]
                                 : xa[half] * ya[half];
        v8f cz = {};
        mid_d[half] = __builtin_amdgcn_wmma_f32_16x16x32_f16(
            false, a, false, wh, (short)0, cz, false, false);
      }

      // ---- spill mid column-major as f16 (one b128 store per half) ...
      #pragma unroll
      for (int half = 0; half < 2; ++half) {
        const int n     = lane & 15;
        const int mbase = half * 16 + ((lane < 16) ? 0 : 8);
        v8h pk;
        #pragma unroll
        for (int i = 0; i < 8; ++i) pk[i] = (_Float16)mid_d[half][i];
        *(v8h*)&L.midT[n][mbase] = pk;
      }
      // ... and re-gather as B operand (two b128 loads, table order in k).
      v16h b;
      {
        const int n   = lane & 15;
        const int kb2 = (lane >> 4) * 8;
        const v8h blo = *(const v8h*)&L.midT[n][kb2];
        const v8h bhi = *(const v8h*)&L.midT[n][kb2 + 16];
        b = SHUF16(blo, bhi);
      }

      // ---- vertical pass: out(16x16) = Wv(16x32) * mid(32x16)
      v8f cz = {};
      out[mp] = __builtin_amdgcn_wmma_f32_16x16x32_f16(
          false, wv, false, b, (short)0, cz, false, false);
    }

    // Per-pixel SSIM on the 8 f32 accumulator elements (mean is layout-agnostic).
    #pragma unroll
    for (int i = 0; i < 8; ++i) {
      const float mu1 = out[0][i], mu2 = out[1][i];
      const float s1  = out[2][i] - mu1 * mu1;
      const float s2  = out[3][i] - mu2 * mu2;
      const float s12 = out[4][i] - mu1 * mu2;
      const float num = (2.0f * mu1 * mu2 + SSIM_C1) * (2.0f * s12 + SSIM_C2);
      const float den = (mu1 * mu1 + mu2 * mu2 + SSIM_C1) * (s1 + s2 + SSIM_C2);
      acc += num / den;
    }
  }

  // wave32 shuffle reduction, then block partial (overwritten every call).
  #pragma unroll
  for (int o = 16; o > 0; o >>= 1) acc += __shfl_down(acc, o, 32);
  if (lane == 0) wsum[wave] = acc;
  __syncthreads();
  if (threadIdx.x == 0) {
    float s = 0.0f;
    #pragma unroll
    for (int w = 0; w < WPG; ++w) s += wsum[w];
    partial[blockIdx.x] = s;
  }
}

__global__ __launch_bounds__(256, 1)
void ssim_reduce(const float* __restrict__ partial, float* __restrict__ out)
{
  __shared__ float s[256];
  float a = 0.0f;
  for (int i = threadIdx.x; i < NWG; i += 256) a += partial[i];
  s[threadIdx.x] = a;
  __syncthreads();
  #pragma unroll
  for (int o = 128; o > 0; o >>= 1) {
    if (threadIdx.x < o) s[threadIdx.x] += s[threadIdx.x + o];
    __syncthreads();
  }
  if (threadIdx.x == 0) {
    const float inv = 1.0f / (float)((size_t)PLANES * IMG_H * IMG_W);
    out[0] = 1.0f - s[0] * inv;
  }
}

extern "C" void kernel_launch(void* const* d_in, const int* in_sizes, int n_in,
                              void* d_out, int out_size, void* d_ws, size_t ws_size,
                              hipStream_t stream)
{
  (void)in_sizes; (void)n_in; (void)out_size; (void)ws_size;
  const float* gen = (const float*)d_in[0];   // [32,3,512,512] f32
  const float* ref = (const float*)d_in[1];   // [32,3,512,512] f32
  const float* win = (const float*)d_in[2];   // [3,1,11,11]    f32
  float* partial   = (float*)d_ws;            // NWG f32 partials (8 KB)
  float* out       = (float*)d_out;           // scalar f32

  ssim_tiles <<<NWG, 256, 0, stream>>>(gen, ref, win, partial);
  ssim_reduce<<<1,   256, 0, stream>>>(partial, out);
}
